// CausalSelfAttention_10926396801511
// MI455X (gfx1250) — compile-verified, matrix operand order checked by reading, not run on hardware
//
#include <hip/hip_runtime.h>

#define N_EMBED 1024
#define N_HEAD  16
#define HD      64
#define BB      4
#define TT      2048
#define BT      (BB * TT)   // 8192 rows

typedef __bf16 v8bf  __attribute__((ext_vector_type(8)));
typedef __bf16 v16bf __attribute__((ext_vector_type(16)));
typedef float  v8f   __attribute__((ext_vector_type(8)));
typedef float  f32x4 __attribute__((ext_vector_type(4)));
typedef int    v4i   __attribute__((ext_vector_type(4)));

// ---- CDNA5 16-bit A/B fragment layout --------------------------------------
// Per lane: elements 0..7 = K {kb8..kb8+7}, elements 8..15 = K {16+kb8..23+kb8},
// kb8 = (lane>>4)*8  => two contiguous 16-byte runs.
__device__ __forceinline__ v16bf frag_bf16(const __bf16* base, int kb8) {
    union { v16bf v; v8bf h[2]; } u;
    u.h[0] = *(const v8bf*)(base + kb8);
    u.h[1] = *(const v8bf*)(base + 16 + kb8);
    return u.v;
}

#define WMMA_BF16(a, b, c) \
    __builtin_amdgcn_wmma_f32_16x16x32_bf16(false, (a), false, (b), (short)0, (c), false, false)

// ---- async global->LDS staging (gfx1250 ASYNCcnt path, guarded) -------------
#if __has_builtin(__builtin_amdgcn_global_load_async_to_lds_b128) && \
    __has_builtin(__builtin_amdgcn_s_wait_asynccnt)
#define HAS_ASYNC_LDS 1
#else
#define HAS_ASYNC_LDS 0
#endif

__device__ __forceinline__ void copy16_g2l(const __bf16* g, __bf16* l) {
#if HAS_ASYNC_LDS
    __builtin_amdgcn_global_load_async_to_lds_b128(
        (__attribute__((address_space(1))) v4i*)g,
        (__attribute__((address_space(3))) v4i*)l, 0, 0);
#else
    *(v8bf*)l = *(const v8bf*)g;
#endif
}

// ---- kernel 0: one-shot f32 -> bf16 conversion ------------------------------
__global__ void cvt_bf16(const float* __restrict__ src, __bf16* __restrict__ dst, int n) {
    const int i = (blockIdx.x * blockDim.x + threadIdx.x) * 8;
    if (i >= n) return;
    f32x4 a = *(const f32x4*)(src + i);
    f32x4 b = *(const f32x4*)(src + i + 4);
    v8bf r;
    r[0] = (__bf16)a[0]; r[1] = (__bf16)a[1]; r[2] = (__bf16)a[2]; r[3] = (__bf16)a[3];
    r[4] = (__bf16)b[0]; r[5] = (__bf16)b[1]; r[6] = (__bf16)b[2]; r[7] = (__bf16)b[3];
    *(v8bf*)(dst + i) = r;
}

// ---- kernel 1: projection GEMM  out = x @ W^T + bias  (bf16 in, bf16 out) ---
// One wave computes a 32x64 output panel: 2 A-frags x 4 B-frags -> 8 WMMAs/step,
// each B-fragment reused across two accumulator rows.
__global__ void qkv_gemm(const __bf16* __restrict__ xb, const __bf16* __restrict__ Wb,
                         const float* __restrict__ bias, __bf16* __restrict__ dst,
                         int transposed) {
    const int lane = threadIdx.x;
    const int l15 = lane & 15, grp = lane >> 4, kb8 = grp * 8;
    const int m0 = blockIdx.x * 32, o0 = blockIdx.y * 64;

    const __bf16* arow0 = xb + (size_t)(m0 + l15) * N_EMBED;
    const __bf16* arow1 = xb + (size_t)(m0 + 16 + l15) * N_EMBED;
    const __bf16* brow[4];
#pragma unroll
    for (int j = 0; j < 4; ++j)
        brow[j] = Wb + (size_t)(o0 + 16 * j + l15) * N_EMBED;   // row o of W = col o of W^T

    v8f acc[2][4] = {};
    for (int k0 = 0; k0 < N_EMBED; k0 += 32) {
        v16bf a0 = frag_bf16(arow0 + k0, kb8);
        v16bf a1 = frag_bf16(arow1 + k0, kb8);
#pragma unroll
        for (int j = 0; j < 4; ++j) {
            v16bf bf = frag_bf16(brow[j] + k0, kb8);
            acc[0][j] = WMMA_BF16(a0, bf, acc[0][j]);
            acc[1][j] = WMMA_BF16(a1, bf, acc[1][j]);
        }
    }

#pragma unroll
    for (int mi = 0; mi < 2; ++mi)
#pragma unroll
        for (int j = 0; j < 4; ++j) {
            const int colg = o0 + 16 * j + l15;
            const int h = colg >> 6, d = colg & 63;
            const float bv = bias[colg];
#pragma unroll
            for (int r = 0; r < 8; ++r) {
                const int rowg = m0 + 16 * mi + r + 8 * grp;
                const int bidx = rowg >> 11, t = rowg & (TT - 1);
                const float val = acc[mi][j][r] + bv;
                const size_t idx = transposed
                    ? (((size_t)bidx * N_HEAD + h) * HD + d) * TT + t
                    : (((size_t)bidx * N_HEAD + h) * TT + t) * HD + d;
                dst[idx] = (__bf16)val;
            }
        }
}

// ---- kernel 2: flash attention ----------------------------------------------
// 4 waves/block over a 64-row query panel. K/V tiles double-buffered in LDS:
// async copies of tile i+1 are issued before computing tile i, overlapped via
// s_wait_asynccnt(4) (async ops complete in order; 4 newly-issued remain).
__device__ __forceinline__ void stage_tiles(const __bf16* K, const __bf16* V,
                                            int kbase, int tid,
                                            __bf16 (*Kt)[64], __bf16 (*Vl)[32]) {
#pragma unroll
    for (int i = 0; i < 2; ++i) {
        const int idx = tid + 128 * i;
        const int kr = idx >> 3, kc = (idx & 7) * 8;
        copy16_g2l(K + (size_t)(kbase + kr) * HD + kc, &Kt[kr][kc]);
        const int vr = idx >> 2, vc = (idx & 3) * 8;
        copy16_g2l(V + (size_t)vr * TT + kbase + vc, &Vl[vr][vc]);
    }
}

__global__ void flash_attn(const __bf16* __restrict__ Qb, const __bf16* __restrict__ Kb,
                           const __bf16* __restrict__ Vt, __bf16* __restrict__ Yb) {
    __shared__ __align__(16) __bf16 Kt[2][32][64];     // [buf][k-row][d]
    __shared__ __align__(16) __bf16 Vl[2][64][32];     // [buf][d][k-row]
    __shared__ __align__(16) __bf16 ptile[4][16][32];  // per-wave prob relayout

    const int tid  = threadIdx.x;
    const int w    = tid >> 5;          // wave id 0..3
    const int lane = tid & 31;
    const int l15 = lane & 15, grp = lane >> 4, kb8 = grp * 8;

    const int t0blk = blockIdx.x * 64;
    const int t0w   = t0blk + 16 * w;   // this wave's query rows
    const int bh = blockIdx.y;
    const int b = bh >> 4, h = bh & 15;

    const __bf16* Q = Qb + (size_t)bh * TT * HD;
    const __bf16* K = Kb + (size_t)bh * TT * HD;
    const __bf16* V = Vt + (size_t)bh * HD * TT;   // [d][t]

    const __bf16* qrow = Q + (size_t)(t0w + l15) * HD;
    v16bf qf0 = frag_bf16(qrow, kb8);
    v16bf qf1 = frag_bf16(qrow + 32, kb8);

    float mrow[8], lrow[8];
    v8f acc[4] = {};
#pragma unroll
    for (int r = 0; r < 8; ++r) { mrow[r] = -__builtin_inff(); lrow[r] = 0.0f; }
    const float NEG_INF = -__builtin_inff();

    stage_tiles(K, V, 0, tid, Kt[0], Vl[0]);    // prologue prefetch
    int cur = 0;

    for (int kbase = 0; kbase < t0blk + 64; kbase += 32) {
        // ---- issue next tile's async copies, then wait for current tile ----
        if (kbase + 32 < t0blk + 64) {
            stage_tiles(K, V, kbase + 32, tid, Kt[cur ^ 1], Vl[cur ^ 1]);
#if HAS_ASYNC_LDS
            __builtin_amdgcn_s_wait_asynccnt(4);
#endif
        } else {
#if HAS_ASYNC_LDS
            __builtin_amdgcn_s_wait_asynccnt(0);
#endif
        }
        __syncthreads();   // current tile visible block-wide

        if (kbase < t0w + 16) {   // wave-uniform causal skip (EXEC stays full)
            const __bf16* k0row = &Kt[cur][l15][0];
            const __bf16* k1row = &Kt[cur][16 + l15][0];
            v8f s0 = {}, s1 = {};
            s0 = WMMA_BF16(qf0, frag_bf16(k0row,      kb8), s0);
            s0 = WMMA_BF16(qf1, frag_bf16(k0row + 32, kb8), s0);
            s1 = WMMA_BF16(qf0, frag_bf16(k1row,      kb8), s1);
            s1 = WMMA_BF16(qf1, frag_bf16(k1row + 32, kb8), s1);

            float p0[8], p1[8], scf[8];
#pragma unroll
            for (int r = 0; r < 8; ++r) {
                const int trow = t0w + r + 8 * grp;
                float a0 = s0[r] * 0.125f;          // 1/sqrt(64)
                float a1 = s1[r] * 0.125f;
                if (kbase + l15 > trow)      a0 = NEG_INF;
                if (kbase + 16 + l15 > trow) a1 = NEG_INF;
                float vm = fmaxf(a0, a1);
                vm = fmaxf(vm, __shfl_xor(vm, 1));
                vm = fmaxf(vm, __shfl_xor(vm, 2));
                vm = fmaxf(vm, __shfl_xor(vm, 4));
                vm = fmaxf(vm, __shfl_xor(vm, 8));
                const float mnew = fmaxf(mrow[r], vm);
                const float sc = __expf(mrow[r] - mnew);
                const float e0 = __expf(a0 - mnew);
                const float e1 = __expf(a1 - mnew);
                float ps = e0 + e1;
                ps += __shfl_xor(ps, 1);
                ps += __shfl_xor(ps, 2);
                ps += __shfl_xor(ps, 4);
                ps += __shfl_xor(ps, 8);
                lrow[r] = lrow[r] * sc + ps;
                mrow[r] = mnew;
                scf[r] = sc; p0[r] = e0; p1[r] = e1;
            }
#pragma unroll
            for (int dt = 0; dt < 4; ++dt)
#pragma unroll
                for (int r = 0; r < 8; ++r) acc[dt][r] *= scf[r];

            // D-layout -> A-layout transpose within this wave's ptile slot
#pragma unroll
            for (int r = 0; r < 8; ++r) {
                const int row = r + 8 * grp;
                ptile[w][row][l15]      = (__bf16)p0[r];
                ptile[w][row][16 + l15] = (__bf16)p1[r];
            }
            __builtin_amdgcn_fence(__ATOMIC_ACQ_REL, "wavefront");
            v16bf pf = frag_bf16(&ptile[w][l15][0], kb8);

#pragma unroll
            for (int dt = 0; dt < 4; ++dt) {
                v16bf vf = frag_bf16(&Vl[cur][dt * 16 + l15][0], kb8);
                acc[dt] = WMMA_BF16(pf, vf, acc[dt]);
            }
        }
        __syncthreads();   // all waves done reading before buffer is refilled
        cur ^= 1;
    }

    // normalize and store merged-head output [B,T,C] bf16
#pragma unroll
    for (int dt = 0; dt < 4; ++dt)
#pragma unroll
        for (int r = 0; r < 8; ++r) {
            const int t = t0w + r + 8 * grp;
            const float val = acc[dt][r] / lrow[r];
            Yb[((size_t)b * TT + t) * N_EMBED + h * HD + dt * 16 + l15] = (__bf16)val;
        }
}

// ---- kernel 3: output projection  out = Y @ Wp^T + bp  -> f32 ---------------
__global__ void out_proj(const __bf16* __restrict__ Yb, const __bf16* __restrict__ Wpb,
                         const float* __restrict__ bp, float* __restrict__ out) {
    const int lane = threadIdx.x;
    const int l15 = lane & 15, grp = lane >> 4, kb8 = grp * 8;
    const int m0 = blockIdx.x * 32, o0 = blockIdx.y * 64;

    const __bf16* arow0 = Yb + (size_t)(m0 + l15) * N_EMBED;
    const __bf16* arow1 = Yb + (size_t)(m0 + 16 + l15) * N_EMBED;
    const __bf16* brow[4];
#pragma unroll
    for (int j = 0; j < 4; ++j)
        brow[j] = Wpb + (size_t)(o0 + 16 * j + l15) * N_EMBED;

    v8f acc[2][4] = {};
    for (int k0 = 0; k0 < N_EMBED; k0 += 32) {
        v16bf a0 = frag_bf16(arow0 + k0, kb8);
        v16bf a1 = frag_bf16(arow1 + k0, kb8);
#pragma unroll
        for (int j = 0; j < 4; ++j) {
            v16bf bf = frag_bf16(brow[j] + k0, kb8);
            acc[0][j] = WMMA_BF16(a0, bf, acc[0][j]);
            acc[1][j] = WMMA_BF16(a1, bf, acc[1][j]);
        }
    }

#pragma unroll
    for (int mi = 0; mi < 2; ++mi)
#pragma unroll
        for (int j = 0; j < 4; ++j) {
            const int colg = o0 + 16 * j + l15;
            const float bv = bp[colg];
#pragma unroll
            for (int r = 0; r < 8; ++r) {
                const int rowg = m0 + 16 * mi + r + 8 * grp;
                out[(size_t)rowg * N_EMBED + colg] = acc[mi][j][r] + bv;
            }
        }
}

// ---- launch -----------------------------------------------------------------
extern "C" void kernel_launch(void* const* d_in, const int* in_sizes, int n_in,
                              void* d_out, int out_size, void* d_ws, size_t ws_size,
                              hipStream_t stream) {
    const float* x  = (const float*)d_in[0];
    const float* Wk = (const float*)d_in[1];
    const float* bk = (const float*)d_in[2];
    const float* Wq = (const float*)d_in[3];
    const float* bq = (const float*)d_in[4];
    const float* Wv = (const float*)d_in[5];
    const float* bv = (const float*)d_in[6];
    const float* Wp = (const float*)d_in[7];
    const float* bp = (const float*)d_in[8];

    char* ws = (char*)d_ws;
    const size_t XB  = (size_t)BT * N_EMBED * sizeof(__bf16);              // 16 MB
    const size_t WB  = (size_t)N_EMBED * N_EMBED * sizeof(__bf16);         //  2 MB
    const size_t SZ  = (size_t)BB * N_HEAD * TT * HD * sizeof(__bf16);     // 16 MB

    __bf16* xb  = (__bf16*)(ws);
    __bf16* Wkb = (__bf16*)(ws + XB);
    __bf16* Wqb = (__bf16*)(ws + XB + WB);
    __bf16* Wvb = (__bf16*)(ws + XB + 2 * WB);
    __bf16* Wpb = (__bf16*)(ws + XB + 3 * WB);
    __bf16* Qb  = (__bf16*)(ws + XB + 4 * WB);
    __bf16* Kb  = (__bf16*)(ws + XB + 4 * WB + SZ);
    __bf16* Vt  = (__bf16*)(ws + XB + 4 * WB + 2 * SZ);
    __bf16* Yb  = (__bf16*)(ws + XB + 4 * WB + 3 * SZ);

    const int nX = BT * N_EMBED, nW = N_EMBED * N_EMBED;
    cvt_bf16<<<nX / 8 / 256, 256, 0, stream>>>(x,  xb,  nX);
    cvt_bf16<<<nW / 8 / 256, 256, 0, stream>>>(Wk, Wkb, nW);
    cvt_bf16<<<nW / 8 / 256, 256, 0, stream>>>(Wq, Wqb, nW);
    cvt_bf16<<<nW / 8 / 256, 256, 0, stream>>>(Wv, Wvb, nW);
    cvt_bf16<<<nW / 8 / 256, 256, 0, stream>>>(Wp, Wpb, nW);

    dim3 gGemm(BT / 32, N_EMBED / 64);
    qkv_gemm<<<gGemm, 32, 0, stream>>>(xb, Wqb, bq, Qb, 0);
    qkv_gemm<<<gGemm, 32, 0, stream>>>(xb, Wkb, bk, Kb, 0);
    qkv_gemm<<<gGemm, 32, 0, stream>>>(xb, Wvb, bv, Vt, 1);

    dim3 gAttn(TT / 64, BB * N_HEAD);
    flash_attn<<<gAttn, 128, 0, stream>>>(Qb, Kb, Vt, Yb);

    out_proj<<<gGemm, 32, 0, stream>>>(Yb, Wpb, bp, (float*)d_out);
}
